// ExtractorMLP_32126355374952
// MI455X (gfx1250) — matrix operand order repaired and checked by reading, not compiled, not run on hardware
//
#include <hip/hip_runtime.h>
#include <hip/hip_bf16.h>

// CDNA5 / gfx1250, wave32.
typedef __attribute__((ext_vector_type(16))) _Float16 v16h;
typedef __attribute__((ext_vector_type(8)))  float    v8f;

#define HD  256   // hidden dim (fixed by the model)
#define WPB 4     // waves (16-row blocks) per block in the MLP kernel

struct h4 { _Float16 x, y, z, w; };          // 8-byte packed f16 quad

union FragU { v16h v; int4 q[2]; };

__device__ __forceinline__ v16h make_frag(int4 q0, int4 q1) {
  FragU u; u.q[0] = q0; u.q[1] = q1; return u.v;
}

// A fragment (16x32 f16) from a row-major [16][256] f16 LDS stage.
// ISA layout: lane l holds row (l&15); lanes 0-15: K {kb*32+0..7, +16..23},
// lanes 16-31: K {kb*32+8..15, +24..31}.
__device__ __forceinline__ v16h load_a_lds(const _Float16* stage, int lane, int kb) {
  const int r    = lane & 15;
  const int half = lane >> 4;
  const _Float16* p = stage + r * HD + kb * 32 + half * 8;
  int4 q0 = *(const int4*)(p);        // ds_load_b128
  int4 q1 = *(const int4*)(p + 16);   // ds_load_b128
  return make_frag(q0, q1);
}

// B fragment (32x16 f16): B[k,n] = W[n,k], W row-major f16 [256][256].
// ISA layout: lanes 0-15 hold col n=(lane&15), K {kb*32+0..7, +16..23};
// lanes 16-31 hold K {kb*32+8..15, +24..31}.  -> two 16B chunks of W row n.
__device__ __forceinline__ v16h load_b_w(const _Float16* __restrict__ W,
                                         int lane, int nb, int kb) {
  const int n    = (nb << 4) + (lane & 15);
  const int half = lane >> 4;
  const _Float16* p = W + n * HD + kb * 32 + half * 8;
  int4 q0 = *(const int4*)(p);        // global_load_b128
  int4 q1 = *(const int4*)(p + 16);   // global_load_b128
  return make_frag(q0, q1);
}

// ---------------------------------------------------------------------------
// Convert the two 256x256 f32 weight matrices to f16 (row-major, no transpose
// needed: B-operand wants W[n,k] which is exactly a row of W).
__global__ __launch_bounds__(256) void cvt_weights_kernel(
    const float* __restrict__ W1, const float* __restrict__ W2,
    _Float16* __restrict__ Wh1, _Float16* __restrict__ Wh2) {
  const int i = (blockIdx.x * 256 + threadIdx.x) * 4;   // 65536 elems total
  float4 a = *(const float4*)(W1 + i);
  float4 b = *(const float4*)(W2 + i);
  h4 ha; ha.x = (_Float16)a.x; ha.y = (_Float16)a.y; ha.z = (_Float16)a.z; ha.w = (_Float16)a.w;
  h4 hb; hb.x = (_Float16)b.x; hb.y = (_Float16)b.y; hb.z = (_Float16)b.z; hb.w = (_Float16)b.w;
  *(h4*)(Wh1 + i) = ha;
  *(h4*)(Wh2 + i) = hb;
}

// ---------------------------------------------------------------------------
// Per-node MLP + L2-normalization.  One wave per 16-row block.
// Output G layout (f16): addr = blk*4096 + t*256 + r*32 + lane, which encodes
// element (m,n) with m = r + (lane>=16 ? 8:0), n = (lane&15) + 16*t  — i.e. the
// native WMMA f32 C/D layout, so stores are fully coalesced 64B segments.
__global__ __launch_bounds__(32 * WPB) void mlp_norm_kernel(
    const float* __restrict__ emb,
    const float* __restrict__ b1, const float* __restrict__ b2,
    const _Float16* __restrict__ Wh1, const _Float16* __restrict__ Wh2,
    _Float16* __restrict__ G, int nrb, int nrows) {
  __shared__ __align__(16) _Float16 stage_s[WPB][16 * HD];   // 8KB per wave
  const int wave = threadIdx.x >> 5;
  const int lane = threadIdx.x & 31;
  const int rb   = blockIdx.x * WPB + wave;
  if (rb >= nrb) return;                 // wave-uniform: EXEC stays all-ones
  _Float16* stage = stage_s[wave];

  // ---- stage the 16 input rows (f32 -> f16) into LDS, row-major ----------
  {
    const int r0 = rb * 16;
    for (int it = 0; it < 32; ++it) {
      int idx = it * 128 + lane * 4;                 // 0..4095
      int row = r0 + (idx >> 8);
      if (row >= nrows) row = nrows - 1;             // tail clamp (harmless)
      const float4 v = *(const float4*)(emb + (size_t)row * HD + (idx & 255));
      h4 h; h.x = (_Float16)v.x; h.y = (_Float16)v.y;
            h.z = (_Float16)v.z; h.w = (_Float16)v.w;
      *(h4*)(stage + idx) = h;                        // ds_store_b64
    }
  }

  v8f acc[16];
  const v8f vzero = {};
  const int nlo = lane & 15;
  const int mhi = (lane >> 4) * 8;

  // ================= layer 1: h = relu(x @ W1^T + b1) =====================
  #pragma unroll
  for (int t = 0; t < 16; ++t) acc[t] = vzero;
  for (int kb = 0; kb < 8; ++kb) {
    v16h a = load_a_lds(stage, lane, kb);
    #pragma unroll
    for (int t = 0; t < 16; ++t) {
      v16h b = load_b_w(Wh1, lane, t, kb);
      acc[t] = __builtin_amdgcn_wmma_f32_16x16x32_f16(
          false, a, false, b, (short)0, acc[t], false, false);
    }
  }
  #pragma unroll
  for (int t = 0; t < 16; ++t) {
    const float bias = b1[nlo + 16 * t];
    #pragma unroll
    for (int r = 0; r < 8; ++r) {
      float x = acc[t][r] + bias;
      x = fmaxf(x, 0.0f);
      stage[(r + mhi) * HD + (nlo + 16 * t)] = (_Float16)x;   // ds_store_b16
    }
  }

  // ================= layer 2: g = h @ W2^T + b2 ===========================
  #pragma unroll
  for (int t = 0; t < 16; ++t) acc[t] = vzero;
  for (int kb = 0; kb < 8; ++kb) {
    v16h a = load_a_lds(stage, lane, kb);
    #pragma unroll
    for (int t = 0; t < 16; ++t) {
      v16h b = load_b_w(Wh2, lane, t, kb);
      acc[t] = __builtin_amdgcn_wmma_f32_16x16x32_f16(
          false, a, false, b, (short)0, acc[t], false, false);
    }
  }

  // ---- bias, row L2 norm (clamped at eps), normalize, store f16 ----------
  {
    float ss[8];
    #pragma unroll
    for (int r = 0; r < 8; ++r) ss[r] = 0.0f;
    #pragma unroll
    for (int t = 0; t < 16; ++t) {
      const float bias = b2[nlo + 16 * t];
      #pragma unroll
      for (int r = 0; r < 8; ++r) {
        float x = acc[t][r] + bias;
        acc[t][r] = x;
        ss[r] += x * x;
      }
    }
    float inv[8];
    #pragma unroll
    for (int r = 0; r < 8; ++r) {
      float s = ss[r];                       // rows live in 16-lane halves
      s += __shfl_xor(s, 1, 32);
      s += __shfl_xor(s, 2, 32);
      s += __shfl_xor(s, 4, 32);
      s += __shfl_xor(s, 8, 32);
      inv[r] = 1.0f / fmaxf(sqrtf(s), 1e-8f);   // torch CosineSimilarity eps
    }
    _Float16* gdst = G + (size_t)rb * (16 * HD);
    #pragma unroll
    for (int t = 0; t < 16; ++t) {
      #pragma unroll
      for (int r = 0; r < 8; ++r) {
        gdst[t * 256 + r * 32 + lane] = (_Float16)(acc[t][r] * inv[r]);
      }
    }
  }
}

// ---------------------------------------------------------------------------
// Per-edge cosine = dot of the two pre-normalized rows.  One wave per edge;
// each lane loads one aligned 16B chunk of each row (tiled-layout address).
__global__ __launch_bounds__(256) void edge_dot_kernel(
    const int* __restrict__ ei, const _Float16* __restrict__ G,
    float* __restrict__ out, int E) {
  const int lane = threadIdx.x & 31;
  const int e = (int)((blockIdx.x * blockDim.x + threadIdx.x) >> 5);
  if (e >= E) return;
  const int mc = ei[e];                 // col
  const int mr = ei[(size_t)E + e];     // row
  const int t    = lane >> 1;
  const int off8 = (lane & 1) << 3;

  const size_t ac = (size_t)(mc >> 4) * 4096 + (size_t)t * 256 +
                    ((mc & 7) << 5) + ((mc & 8) << 1) + off8;
  const size_t ar = (size_t)(mr >> 4) * 4096 + (size_t)t * 256 +
                    ((mr & 7) << 5) + ((mr & 8) << 1) + off8;
  const int4 qa = *(const int4*)(G + ac);   // global_load_b128
  const int4 qb = *(const int4*)(G + ar);   // global_load_b128
  const _Float16* ha = (const _Float16*)&qa;
  const _Float16* hb = (const _Float16*)&qb;

  float s = 0.0f;
  #pragma unroll
  for (int i = 0; i < 8; ++i) s += (float)ha[i] * (float)hb[i];
  s += __shfl_xor(s, 1, 32);
  s += __shfl_xor(s, 2, 32);
  s += __shfl_xor(s, 4, 32);
  s += __shfl_xor(s, 8, 32);
  s += __shfl_xor(s, 16, 32);
  if (lane == 0) out[e] = s;
}

// ---------------------------------------------------------------------------
extern "C" void kernel_launch(void* const* d_in, const int* in_sizes, int n_in,
                              void* d_out, int out_size, void* d_ws, size_t ws_size,
                              hipStream_t stream) {
  const float* emb = (const float*)d_in[0];
  const int*   ei  = (const int*)d_in[1];   // edge_index (harness integer ABI)
  const float* W1  = (const float*)d_in[2];
  const float* b1  = (const float*)d_in[3];
  const float* W2  = (const float*)d_in[4];
  const float* b2  = (const float*)d_in[5];
  float* out = (float*)d_out;

  const int N = in_sizes[0] / HD;      // 50000 nodes
  const int E = in_sizes[1] / 2;       // 300000 edges

  char* ws = (char*)d_ws;
  _Float16* Wh1 = (_Float16*)(ws);                  // 128 KB
  _Float16* Wh2 = (_Float16*)(ws + 131072);         // 128 KB
  _Float16* G   = (_Float16*)(ws + 262144);         // nrb*16*256 f16 (~25.6 MB)

  // 1) weights f32 -> f16 (65536 elems, 4 per thread)
  cvt_weights_kernel<<<64, 256, 0, stream>>>(W1, W2, Wh1, Wh2);

  // 2) per-node MLP + normalization
  const int nrb = (N + 15) / 16;
  mlp_norm_kernel<<<(nrb + WPB - 1) / WPB, 32 * WPB, 0, stream>>>(
      emb, b1, b2, Wh1, Wh2, G, nrb, N);

  // 3) per-edge cosine (one wave per edge, 8 waves per block)
  edge_dot_kernel<<<(E + 7) / 8, 256, 0, stream>>>(ei, G, out, E);
}
